// ResidualBlock_27934467293189
// MI455X (gfx1250) — matrix operand order repaired and checked by reading, not compiled, not run on hardware
//
#include <hip/hip_runtime.h>

// ---------------- types / constants ----------------
typedef _Float16 v16h __attribute__((ext_vector_type(16)));
typedef _Float16 h2v  __attribute__((ext_vector_type(2)));
typedef float    v8f  __attribute__((ext_vector_type(8)));

constexpr int Bn = 4, C = 128, H = 96, W = 96;
constexpr int HW = H * W;              // 9216
constexpr int KK = 9;
constexpr int KDIM = C * KK;           // 1152
constexpr int NKB = KDIM / 32;         // 36 K-blocks of 32
constexpr float EPS = 1e-5f;
constexpr float SLOPE = 0.2f;

#define DEVFN static __device__ __forceinline__

// ---------------- CDNA5 async global->LDS copy (ISA 08_async_tensor §4) -------
DEVFN void async_copy_b128(uint32_t lds_off, const void* gaddr) {
  asm volatile("global_load_async_to_lds_b128 %0, %1, off"
               :: "v"(lds_off), "v"(gaddr) : "memory");
}
DEVFN void wait_async0() {
  asm volatile("s_wait_asynccnt 0" ::: "memory");
}

// ---------------- WMMA fragment index helpers (per ISA 7.12.2) ----------------
// 16-bit A 16x32, this code's contiguous k order (not the ISA table's): lane<16 -> K 0..15; lane>=16 -> K 16..31 (matches B)
DEVFN int a_koff(int lane, int j) {
  return ((lane & 16) ? 16 : 0) + (j << 1);
}

DEVFN v8f vzero8() {
  v8f z;
#pragma unroll
  for (int e = 0; e < 8; ++e) z[e] = 0.f;
  return z;
}

// B fragment from an LDS tile laid out [N][32] f16 (row = 64B).
// 32x16 B: lanes 0-15 hold K=0..15, lanes 16-31 hold K=16..31 (pairs per VGPR).
DEVFN v16h lds_b_frag(const _Float16* tile, int n, int lane) {
  const _Float16* row = tile + n * 32 + ((lane & 16) ? 16 : 0);
  v16h r;
#pragma unroll
  for (int j = 0; j < 8; ++j) {
    h2v p = *(const h2v*)(row + 2 * j);
    r[2 * j] = p.x; r[2 * j + 1] = p.y;
  }
  return r;
}

// Stage one K-block weight tile wb[N=NT*16][k 32] into LDS via async DMA.
// Each 16B chunk: chunk c -> n = c/4, part = c%4.
template <int NT>
DEVFN void stage_b(const _Float16* __restrict__ wb, int kbase,
                   _Float16* buf, int tid) {
  constexpr int CH = NT * 16 * 4;     // 16B chunks in tile
#pragma unroll 1
  for (int c = tid; c < CH; c += 256) {
    int n = c >> 2, part = c & 3;
    const void* g = wb + (size_t)n * KDIM + kbase + part * 8;
    uint32_t l = (uint32_t)(uintptr_t)(buf + n * 32 + part * 8);
    async_copy_b128(l, g);
  }
}

// ---------------- weight packing: OIHW f32 -> [O][t*128+c] f16 ----------------
__global__ void pack_w(const float* __restrict__ w, _Float16* __restrict__ wb,
                       int O, int Oreal) {
  int idx = blockIdx.x * blockDim.x + threadIdx.x;
  int total = O * KDIM;
  for (; idx < total; idx += gridDim.x * blockDim.x) {
    int o = idx / KDIM, k = idx % KDIM;
    int t = k >> 7, c = k & 127;
    float v = (o < Oreal) ? w[((size_t)(o * C + c)) * KK + t] : 0.f;
    wb[idx] = (_Float16)v;
  }
}

// ---------------- instance norm 1: stats over NCHW planes ----------------
__global__ __launch_bounds__(256) void in1_stats(const float* __restrict__ x,
                                                 float* __restrict__ mean,
                                                 float* __restrict__ rstd) {
  int bc = blockIdx.x;                       // b*C + c
  const float* p = x + (size_t)bc * HW;
  float s = 0.f, s2 = 0.f;
  for (int i = threadIdx.x; i < HW; i += 256) { float v = p[i]; s += v; s2 += v * v; }
  __shared__ float ls[256], lq[256];
  ls[threadIdx.x] = s; lq[threadIdx.x] = s2; __syncthreads();
  for (int off = 128; off > 0; off >>= 1) {
    if ((int)threadIdx.x < off) { ls[threadIdx.x] += ls[threadIdx.x + off]; lq[threadIdx.x] += lq[threadIdx.x + off]; }
    __syncthreads();
  }
  if (threadIdx.x == 0) {
    float m = ls[0] / HW;
    float var = lq[0] / HW - m * m;
    mean[bc] = m; rstd[bc] = rsqrtf(var + EPS);
  }
}

// ---------------- IN1 apply + LReLU + NCHW->NHWC f16 (LDS transpose) ----------
__global__ __launch_bounds__(256) void in1_apply(const float* __restrict__ x,
                                                 const float* __restrict__ mean,
                                                 const float* __restrict__ rstd,
                                                 const float* __restrict__ g,
                                                 const float* __restrict__ be,
                                                 _Float16* __restrict__ h1) {
  int tile = blockIdx.x;
  int pt = tile % (HW / 16); tile /= (HW / 16);
  int ct = tile % (C / 16);  int b = tile / (C / 16);
  int tx = threadIdx.x & 15, ty = threadIdx.x >> 4;
  int c = ct * 16 + ty;
  float v = x[((size_t)(b * C + c)) * HW + pt * 16 + tx];
  float m = mean[b * C + c], r = rstd[b * C + c];
  v = (v - m) * r * g[c] + be[c];
  v = v >= 0.f ? v : SLOPE * v;
  __shared__ float t_[16][17];
  t_[ty][tx] = v; __syncthreads();
  h1[((size_t)(b * HW + pt * 16 + ty)) * C + ct * 16 + tx] = (_Float16)t_[tx][ty];
}

// ---------------- implicit-GEMM 3x3 conv, wave = 16 pixels x (NT*16) ch -------
// Weights double-buffered in LDS via async copies; all 8 waves share them.
template <int NT>
__global__ __launch_bounds__(256) void conv3x3_wmma(const _Float16* __restrict__ act,
                                                    const _Float16* __restrict__ wb,
                                                    const float* __restrict__ bias,
                                                    int nreal,
                                                    float* __restrict__ out,
                                                    int ldout) {
  __shared__ alignas(16) _Float16 bt[2][NT * 16 * 32];
  int tid = threadIdx.x;
  int wave = tid >> 5, lane = tid & 31;
  int tilem = blockIdx.x * 8 + wave;
  int p0 = tilem * 16;
  int b = p0 / HW, rem = p0 % HW, y = rem / W, xb = rem % W;
  int m = lane & 15, n0 = lane & 15;

  v8f acc[NT];
#pragma unroll
  for (int i = 0; i < NT; ++i) acc[i] = vzero8();

  stage_b<NT>(wb, 0, bt[0], tid);

#pragma unroll 1
  for (int kb = 0; kb < NKB; ++kb) {
    // ---- A fragment for this K-block: pure global loads, issued BEFORE the
    // barrier so their latency hides under barrier arrival + async issue.
    int t = kb >> 2, cb = kb & 3;          // k = t*128 + cb*32
    int iy = y + t / 3 - 1;
    int ix = xb + m + t % 3 - 1;
    bool ok = (iy >= 0) && (iy < H) && (ix >= 0) && (ix < W);
    int iyc = ok ? iy : 0, ixc = ok ? ix : 0;
    const _Float16* row = act + ((size_t)(b * H + iyc) * W + ixc) * C + cb * 32;
    v16h a;
#pragma unroll
    for (int j = 0; j < 8; ++j) {
      h2v p = *(const h2v*)(row + a_koff(lane, j));
      a[2 * j]     = ok ? p.x : (_Float16)0.f;
      a[2 * j + 1] = ok ? p.y : (_Float16)0.f;
    }

    wait_async0();
    __syncthreads();                       // tile kb ready; prev reads done
    if (kb + 1 < NKB) stage_b<NT>(wb, (kb + 1) * 32, bt[(kb + 1) & 1], tid);

    // ---- pipelined B fragments: load nt+1 before consuming nt
    const _Float16* tile = bt[kb & 1];
    v16h bf = lds_b_frag(tile, n0, lane);
#pragma unroll
    for (int nt = 0; nt < NT; ++nt) {
      v16h bfn = bf;
      if (nt + 1 < NT) bfn = lds_b_frag(tile, (nt + 1) * 16 + n0, lane);
      acc[nt] = __builtin_amdgcn_wmma_f32_16x16x32_f16(
          false, a, false, bf, (short)0, acc[nt], false, false);
      bf = bfn;
    }
  }

  int mlo = (lane & 16) ? 8 : 0;
#pragma unroll
  for (int nt = 0; nt < NT; ++nt) {
    int n = nt * 16 + n0;
    float bv = (n < nreal) ? bias[n] : 0.f;
#pragma unroll
    for (int r = 0; r < 8; ++r)
      out[((size_t)(p0 + mlo + r)) * ldout + n] = acc[nt][r] + bv;
  }
}

// ---------------- IN2 stats (NHWC) via atomic partial sums --------------------
__global__ __launch_bounds__(128) void in2_stats(const float* __restrict__ h2,
                                                 float* __restrict__ sum,
                                                 float* __restrict__ sumsq) {
  int blk = blockIdx.x;
  int b = blk >> 6, chunk = blk & 63;             // 64 chunks of 144 pixels
  int c = threadIdx.x;
  const float* base = h2 + ((size_t)b * HW + chunk * 144) * C + c;
  float s = 0.f, s2 = 0.f;
  for (int i = 0; i < 144; ++i) { float v = base[(size_t)i * C]; s += v; s2 += v * v; }
  atomicAdd(&sum[b * C + c], s);
  atomicAdd(&sumsq[b * C + c], s2);
}

// ---------------- IN2 apply + LReLU -> h3 NHWC f16 ----------------------------
__global__ void in2_apply(const float* __restrict__ h2,
                          const float* __restrict__ sum,
                          const float* __restrict__ sumsq,
                          const float* __restrict__ g,
                          const float* __restrict__ be,
                          _Float16* __restrict__ h3) {
  int idx = blockIdx.x * blockDim.x + threadIdx.x;
  const int total = Bn * HW * C;
  for (; idx < total; idx += gridDim.x * blockDim.x) {
    int c = idx & 127;
    int b = idx / (HW * C);
    float m = sum[b * C + c] * (1.f / HW);
    float var = sumsq[b * C + c] * (1.f / HW) - m * m;
    float r = rsqrtf(var + EPS);
    float v = (h2[idx] - m) * r * g[c] + be[c];
    h3[idx] = (_Float16)(v >= 0.f ? v : SLOPE * v);
  }
}

// ---------------- modulated deformable conv + residual ------------------------
__global__ __launch_bounds__(256) void deform_wmma(const _Float16* __restrict__ h3,
                                                   const float* __restrict__ om,
                                                   const _Float16* __restrict__ wb,
                                                   const float* __restrict__ xin,
                                                   float* __restrict__ out) {
  __shared__ alignas(16) _Float16 bt[2][128 * 32];
  int tid = threadIdx.x;
  int wave = tid >> 5, lane = tid & 31;
  int tilem = blockIdx.x * 8 + wave;
  int p0 = tilem * 16;
  int b = p0 / HW, rem = p0 % HW, y = rem / W, xb = rem % W;
  int m = lane & 15, n0 = lane & 15;
  int pix = p0 + m;

  v8f acc[8];
#pragma unroll
  for (int i = 0; i < 8; ++i) acc[i] = vzero8();

  const float* omp = om + (size_t)pix * 32;

  stage_b<8>(wb, 0, bt[0], tid);

  int kb = 0;
#pragma unroll 1
  for (int t = 0; t < 9; ++t) {
    float dy = omp[2 * t], dx = omp[2 * t + 1];
    float mk = 1.f / (1.f + __expf(-omp[18 + t]));
    float py = dy + (float)y - 1.f + (float)(t / 3);
    float px = dx + (float)(xb + m) - 1.f + (float)(t % 3);
    float fy = floorf(py), fx = floorf(px);
    float ly = py - fy, lx = px - fx;
    int y0 = (int)fy, x0 = (int)fx, y1 = y0 + 1, x1 = x0 + 1;
    float w00 = (1.f - ly) * (1.f - lx), w01 = (1.f - ly) * lx;
    float w10 = ly * (1.f - lx),         w11 = ly * lx;
    w00 = (y0 >= 0 && y0 < H && x0 >= 0 && x0 < W) ? w00 * mk : 0.f;
    w01 = (y0 >= 0 && y0 < H && x1 >= 0 && x1 < W) ? w01 * mk : 0.f;
    w10 = (y1 >= 0 && y1 < H && x0 >= 0 && x0 < W) ? w10 * mk : 0.f;
    w11 = (y1 >= 0 && y1 < H && x1 >= 0 && x1 < W) ? w11 * mk : 0.f;
    int y0c = y0 < 0 ? 0 : (y0 > H - 1 ? H - 1 : y0);
    int y1c = y1 < 0 ? 0 : (y1 > H - 1 ? H - 1 : y1);
    int x0c = x0 < 0 ? 0 : (x0 > W - 1 ? W - 1 : x0);
    int x1c = x1 < 0 ? 0 : (x1 > W - 1 ? W - 1 : x1);
    const _Float16* p00 = h3 + ((size_t)(b * H + y0c) * W + x0c) * C;
    const _Float16* p01 = h3 + ((size_t)(b * H + y0c) * W + x1c) * C;
    const _Float16* p10 = h3 + ((size_t)(b * H + y1c) * W + x0c) * C;
    const _Float16* p11 = h3 + ((size_t)(b * H + y1c) * W + x1c) * C;

#pragma unroll 1
    for (int cb = 0; cb < 4; ++cb, ++kb) {
      // ---- build sampled+modulated A fragment BEFORE the barrier (global
      // gathers + VALU only), hiding gather latency under barrier arrival.
      v16h a;
#pragma unroll
      for (int j = 0; j < 8; ++j) {
        int kl = cb * 32 + a_koff(lane, j);
        h2v a00 = *(const h2v*)(p00 + kl);
        h2v a01 = *(const h2v*)(p01 + kl);
        h2v a10 = *(const h2v*)(p10 + kl);
        h2v a11 = *(const h2v*)(p11 + kl);
        float s0 = w00 * (float)a00.x + w01 * (float)a01.x + w10 * (float)a10.x + w11 * (float)a11.x;
        float s1 = w00 * (float)a00.y + w01 * (float)a01.y + w10 * (float)a10.y + w11 * (float)a11.y;
        a[2 * j] = (_Float16)s0; a[2 * j + 1] = (_Float16)s1;
      }

      wait_async0();
      __syncthreads();
      if (kb + 1 < NKB) stage_b<8>(wb, (kb + 1) * 32, bt[(kb + 1) & 1], tid);

      const _Float16* tile = bt[kb & 1];
      v16h bf = lds_b_frag(tile, n0, lane);
#pragma unroll
      for (int nt = 0; nt < 8; ++nt) {
        v16h bfn = bf;
        if (nt + 1 < 8) bfn = lds_b_frag(tile, (nt + 1) * 16 + n0, lane);
        acc[nt] = __builtin_amdgcn_wmma_f32_16x16x32_f16(
            false, a, false, bf, (short)0, acc[nt], false, false);
        bf = bfn;
      }
    }
  }

  int mlo = (lane & 16) ? 8 : 0;
#pragma unroll
  for (int nt = 0; nt < 8; ++nt) {
#pragma unroll
    for (int r = 0; r < 8; ++r) {
      int mm = mlo + r;
      size_t oidx = ((size_t)(b * C + nt * 16 + n0) * H + y) * W + (xb + mm);
      out[oidx] = acc[nt][r] + xin[oidx];
    }
  }
}

// ---------------- launcher ----------------------------------------------------
extern "C" void kernel_launch(void* const* d_in, const int* in_sizes, int n_in,
                              void* d_out, int out_size, void* d_ws, size_t ws_size,
                              hipStream_t stream) {
  const float* x    = (const float*)d_in[0];
  const float* g1   = (const float*)d_in[1];
  const float* b1   = (const float*)d_in[2];
  const float* g2   = (const float*)d_in[3];
  const float* b2   = (const float*)d_in[4];
  const float* w1   = (const float*)d_in[5];
  const float* bc1  = (const float*)d_in[6];
  const float* woff = (const float*)d_in[7];
  const float* boff = (const float*)d_in[8];
  const float* wdef = (const float*)d_in[9];
  float* out = (float*)d_out;

  // workspace layout (bytes)
  char* ws = (char*)d_ws;
  _Float16* h1    = (_Float16*)(ws);                         // 4*9216*128 f16
  _Float16* h3    = (_Float16*)(ws + 9437184);
  float*    h2    = (float*)   (ws + 18874368);              // NHWC f32
  float*    om    = (float*)   (ws + 37748736);              // NHWC f32, 32ch
  _Float16* wb1   = (_Float16*)(ws + 42467328);              // [128][1152]
  _Float16* wbdef = (_Float16*)(ws + 42762240);              // [128][1152]
  _Float16* wboff = (_Float16*)(ws + 43057152);              // [32][1152]
  float*    mean1 = (float*)   (ws + 43130880);              // 512
  float*    rstd1 = (float*)   (ws + 43132928);              // 512
  float*    sum2  = (float*)   (ws + 43134976);              // 512 (+512 sq2)

  // weight packing + stats prep
  pack_w<<<128, 256, 0, stream>>>(w1,   wb1,   128, 128);
  pack_w<<<128, 256, 0, stream>>>(wdef, wbdef, 128, 128);
  pack_w<<<32,  256, 0, stream>>>(woff, wboff, 32,  27);
  in1_stats<<<Bn * C, 256, 0, stream>>>(x, mean1, rstd1);
  hipMemsetAsync(sum2, 0, 2 * Bn * C * sizeof(float), stream);
  float* sq2 = sum2 + Bn * C;

  // IN1 + lrelu + transpose to NHWC f16
  in1_apply<<<Bn * (C / 16) * (HW / 16), 256, 0, stream>>>(x, mean1, rstd1, g1, b1, h1);

  // conv1 (WMMA implicit GEMM, LDS-staged weights): h1 -> h2 (NHWC f32) + bias
  conv3x3_wmma<8><<<(Bn * HW / 16) / 8, 256, 0, stream>>>(h1, wb1, bc1, 128, h2, 128);

  // IN2 + lrelu -> h3 (NHWC f16)
  in2_stats<<<Bn * 64, 128, 0, stream>>>(h2, sum2, sq2);
  in2_apply<<<2048, 256, 0, stream>>>(h2, sum2, sq2, g2, b2, h3);

  // offset/mask conv (WMMA, N padded 27->32): h3 -> om
  conv3x3_wmma<2><<<(Bn * HW / 16) / 8, 256, 0, stream>>>(h3, wboff, boff, 27, om, 32);

  // modulated deformable conv (WMMA) + residual -> out (NCHW f32)
  deform_wmma<<<(Bn * HW / 16) / 8, 256, 0, stream>>>(h3, om, wbdef, x, out);
}
